// KGNNLS_22935125360845
// MI455X (gfx1250) — matrix operand order repaired and checked by reading, not compiled, not run on hardware
//
#include <hip/hip_runtime.h>
#include <hip/hip_bf16.h>

typedef __attribute__((ext_vector_type(16))) _Float16 v16h;
typedef __attribute__((ext_vector_type(8)))  float    v8f;

#define KNB   16     // neighbors per hop
#define DIM   64     // embedding dim
#define WSTR  65     // padded LDS stride for W matrices (bank-conflict-free GEMV)
#define BSTR  66     // padded LDS stride for per-wave X/H tile
#define WAVES 8      // waves (=items) per 256-thread block

__global__ __launch_bounds__(256)
void kgnn_fused_kernel(const int*   __restrict__ u_ids,
                       const int*   __restrict__ i_ids,
                       const int*   __restrict__ adj_entity,
                       const int*   __restrict__ adj_relation,
                       const float* __restrict__ user_emb,
                       const float* __restrict__ entity_emb,
                       const float* __restrict__ relation_emb,
                       const float* __restrict__ W0,
                       const float* __restrict__ b0,
                       const float* __restrict__ W1,
                       const float* __restrict__ b1,
                       float*       __restrict__ out,
                       int B)
{
    __shared__ float sW0[DIM * WSTR];
    __shared__ float sW1[DIM * WSTR];
    __shared__ float sb0[DIM];
    __shared__ float sb1[DIM];
    __shared__ float sbuf[WAVES][KNB * BSTR];  // per-wave 16x64 X tile, then H tile
    __shared__ float sx[WAVES][DIM];           // per-wave 64-vector scratch

    const int tid = threadIdx.x;

    // Cooperative load of weights/biases into LDS (padded stride).
    for (int i = tid; i < DIM * DIM; i += 256) {
        int e = i >> 6, d = i & 63;
        sW0[e * WSTR + d] = W0[i];
        sW1[e * WSTR + d] = W1[i];
    }
    if (tid < DIM) { sb0[tid] = b0[tid]; sb1[tid] = b1[tid]; }
    __syncthreads();

    const int wave = tid >> 5;
    const int lane = tid & 31;
    const int item = blockIdx.x * WAVES + wave;
    if (item >= B) return;  // no barriers after this point

    // ---- user embedding: lane l owns dims 2l, 2l+1 ----
    const int uid = u_ids[item];
    const float2 uev = ((const float2*)(user_emb + (size_t)uid * DIM))[lane];
    const float uex = uev.x, uey = uev.y;

    // ---- srel[r] = dot(ue, relation_emb[r]) / 64, lane r holds relation r (32 relations) ----
    float srel = 0.0f;
    {
        const float* rr = relation_emb + (size_t)lane * DIM;
        for (int d = 0; d < DIM; ++d) {
            float u = (d & 1) ? __shfl(uey, d >> 1, 32) : __shfl(uex, d >> 1, 32);
            srel = fmaf(u, rr[d], srel);
        }
        srel *= (1.0f / 64.0f);
    }

    // ---- hop-1 neighbor indices of the seed (lanes 16..31 duplicate lanes 0..15) ----
    const int e0 = i_ids[item];
    const int kk = lane & 15;
    const int e1k = adj_entity  [(size_t)e0 * KNB + kk];
    const int r1k = adj_relation[(size_t)e0 * KNB + kk];

    // attention over the seed's 16 edges (softmax over 16, duplicated in both lane halves)
    float w1;
    {
        float s = __shfl(srel, r1k, 32);
        float m = s;
        m = fmaxf(m, __shfl_xor(m, 1, 16));
        m = fmaxf(m, __shfl_xor(m, 2, 16));
        m = fmaxf(m, __shfl_xor(m, 4, 16));
        m = fmaxf(m, __shfl_xor(m, 8, 16));
        float ex = __expf(s - m);
        float sum = ex;
        sum += __shfl_xor(sum, 1, 16);
        sum += __shfl_xor(sum, 2, 16);
        sum += __shfl_xor(sum, 4, 16);
        sum += __shfl_xor(sum, 8, 16);
        w1 = ex / (sum * 16.0f);   // includes the mean-over-K factor
    }

    // ---- iteration 0, hop 0: agg over 16 hop-1 rows, then h0 = relu((v0+agg) @ W0^T + b0) ----
    float h0x, h0y;
    {
        float ax = 0.0f, ay = 0.0f;
        for (int k = 0; k < KNB; ++k) {
            float wk = __shfl(w1, k, 32);
            int   ek = __shfl(e1k, k, 32);
            float2 v = ((const float2*)(entity_emb + (size_t)ek * DIM))[lane];
            ax = fmaf(wk, v.x, ax);
            ay = fmaf(wk, v.y, ay);
        }
        float2 v0 = ((const float2*)(entity_emb + (size_t)e0 * DIM))[lane];
        sx[wave][2 * lane]     = v0.x + ax;
        sx[wave][2 * lane + 1] = v0.y + ay;
        float a0 = sb0[2 * lane], a1 = sb0[2 * lane + 1];
        for (int d = 0; d < DIM; ++d) {
            float xv = sx[wave][d];
            a0 = fmaf(xv, sW0[(2 * lane)     * WSTR + d], a0);
            a1 = fmaf(xv, sW0[(2 * lane + 1) * WSTR + d], a1);
        }
        h0x = fmaxf(a0, 0.0f);
        h0y = fmaxf(a1, 0.0f);
    }

    // ---- iteration 0, hop 1: build X = v1 + attn-agg(v2) as a 16x64 tile in LDS ----
    for (int j = 0; j < KNB; ++j) {
        int e1j = __shfl(e1k, j, 32);
        const size_t base = (size_t)e1j * KNB + kk;
        int idx2 = adj_entity[base];
        int r2   = adj_relation[base];
        // prefetch the gather rows for this j (lanes cover all 16 rows)
        __builtin_prefetch(entity_emb + (size_t)idx2 * DIM, 0, 0);

        float s = __shfl(srel, r2, 32);
        float m = s;
        m = fmaxf(m, __shfl_xor(m, 1, 16));
        m = fmaxf(m, __shfl_xor(m, 2, 16));
        m = fmaxf(m, __shfl_xor(m, 4, 16));
        m = fmaxf(m, __shfl_xor(m, 8, 16));
        float ex = __expf(s - m);
        float sum = ex;
        sum += __shfl_xor(sum, 1, 16);
        sum += __shfl_xor(sum, 2, 16);
        sum += __shfl_xor(sum, 4, 16);
        sum += __shfl_xor(sum, 8, 16);
        float w2 = ex / (sum * 16.0f);

        float ax = 0.0f, ay = 0.0f;
        for (int k = 0; k < KNB; ++k) {
            float wk = __shfl(w2, k, 32);
            int   ek = __shfl(idx2, k, 32);
            float2 v = ((const float2*)(entity_emb + (size_t)ek * DIM))[lane];
            ax = fmaf(wk, v.x, ax);
            ay = fmaf(wk, v.y, ay);
        }
        float2 v1 = ((const float2*)(entity_emb + (size_t)e1j * DIM))[lane];
        sbuf[wave][j * BSTR + 2 * lane]     = v1.x + ax;
        sbuf[wave][j * BSTR + 2 * lane + 1] = v1.y + ay;
    }

    // ---- H1 = relu(X @ W0^T + b0) via v_wmma_f32_16x16x32_f16 ----
    const int hi = lane >> 4;   // lane half selects K sub-range per ISA layout
    const int ln = lane & 15;   // A: row M; B/D: column N

    // A-matrix (16x64 f16): per ISA, lane half hi covers K offsets {hi*8 .. hi*8+7} (+16 for v>=4)
    v16h A0, A1;
    for (int v = 0; v < 8; ++v) {
        int kb = ((v < 4) ? 0 : 16) + hi * 8 + 2 * (v & 3);
        A0[2 * v]     = (_Float16)sbuf[wave][ln * BSTR + kb];
        A0[2 * v + 1] = (_Float16)sbuf[wave][ln * BSTR + kb + 1];
        A1[2 * v]     = (_Float16)sbuf[wave][ln * BSTR + 32 + kb];
        A1[2 * v + 1] = (_Float16)sbuf[wave][ln * BSTR + 32 + kb + 1];
    }

    for (int t = 0; t < 4; ++t) {   // N-tiles of the 64 output features
        // B-matrix (32x16 f16): B[k][n] = W0[t*16+n][k]; lane half hi covers K {hi*8 .. hi*8+7} (+16 for v>=4)
        v16h B0v, B1v;
        const float* wrow = &sW0[(t * 16 + ln) * WSTR];
        for (int v = 0; v < 8; ++v) {
            int klo = ((v < 4) ? 0 : 16) + hi * 8 + 2 * (v & 3);
            B0v[2 * v]     = (_Float16)wrow[klo];
            B0v[2 * v + 1] = (_Float16)wrow[klo + 1];
            B1v[2 * v]     = (_Float16)wrow[32 + klo];
            B1v[2 * v + 1] = (_Float16)wrow[32 + klo + 1];
        }
        v8f c;
        float bb = sb0[t * 16 + ln];
        for (int v = 0; v < 8; ++v) c[v] = bb;
        c = __builtin_amdgcn_wmma_f32_16x16x32_f16(false, A0, false, B0v, (short)0, c, false, false);
        c = __builtin_amdgcn_wmma_f32_16x16x32_f16(false, A1, false, B1v, (short)0, c, false, false);
        // D layout: VGPR v, lane -> row m = v + 8*hi, col n = t*16 + ln ; overwrite X tile with H1
        for (int v = 0; v < 8; ++v) {
            int mrow = v + 8 * hi;
            sbuf[wave][mrow * BSTR + t * 16 + ln] = fmaxf(c[v], 0.0f);
        }
    }

    // ---- iteration 1, hop 0: same w1 attention over H1 rows, then tanh((h0+agg) @ W1^T + b1) ----
    float itx, ity;
    {
        float ax = 0.0f, ay = 0.0f;
        for (int k = 0; k < KNB; ++k) {
            float wk = __shfl(w1, k, 32);
            float2 hv = ((const float2*)&sbuf[wave][k * BSTR])[lane];
            ax = fmaf(wk, hv.x, ax);
            ay = fmaf(wk, hv.y, ay);
        }
        sx[wave][2 * lane]     = h0x + ax;
        sx[wave][2 * lane + 1] = h0y + ay;
        float a0 = sb1[2 * lane], a1 = sb1[2 * lane + 1];
        for (int d = 0; d < DIM; ++d) {
            float xv = sx[wave][d];
            a0 = fmaf(xv, sW1[(2 * lane)     * WSTR + d], a0);
            a1 = fmaf(xv, sW1[(2 * lane + 1) * WSTR + d], a1);
        }
        itx = tanhf(a0);
        ity = tanhf(a1);
    }

    // ---- final score: sigmoid(dot(ue, item)) ----
    float dot = uex * itx + uey * ity;
    dot += __shfl_xor(dot, 16, 32);
    dot += __shfl_xor(dot, 8, 32);
    dot += __shfl_xor(dot, 4, 32);
    dot += __shfl_xor(dot, 2, 32);
    dot += __shfl_xor(dot, 1, 32);
    if (lane == 0) out[item] = 1.0f / (1.0f + __expf(-dot));
}

extern "C" void kernel_launch(void* const* d_in, const int* in_sizes, int n_in,
                              void* d_out, int out_size, void* d_ws, size_t ws_size,
                              hipStream_t stream) {
    const int*   u_ids        = (const int*)d_in[0];
    const int*   i_ids        = (const int*)d_in[1];
    const int*   adj_entity   = (const int*)d_in[2];
    const int*   adj_relation = (const int*)d_in[3];
    const float* user_emb     = (const float*)d_in[4];
    const float* entity_emb   = (const float*)d_in[5];
    const float* relation_emb = (const float*)d_in[6];
    const float* W0           = (const float*)d_in[7];
    const float* b0           = (const float*)d_in[8];
    const float* W1           = (const float*)d_in[9];
    const float* b1           = (const float*)d_in[10];
    float* out = (float*)d_out;

    const int B = in_sizes[0];                 // 4096
    const int blocks = (B + WAVES - 1) / WAVES; // 1 wave per item, 8 waves per block
    kgnn_fused_kernel<<<blocks, 256, 0, stream>>>(
        u_ids, i_ids, adj_entity, adj_relation,
        user_emb, entity_emb, relation_emb,
        W0, b0, W1, b1, out, B);
}
